// KVMemNN_88545045775136
// MI455X (gfx1250) — matrix operand order repaired and checked by reading, not compiled, not run on hardware
//
#include <hip/hip_runtime.h>
#include <math.h>

typedef __attribute__((ext_vector_type(16))) _Float16 v16h;
typedef __attribute__((ext_vector_type(8)))  float    v8f;

#define MEMN  1024
#define SENTN 16
#define NEMBD 128
#define HID   40
#define HOPSN 3
#define NANS  20
#define BATCH 64
#define NEGV  -1000000000.0f

// ---------------------------------------------------------------------------
// Kernel 1: per (batch, 16-row memory tile):
//   summed[r,e] = sum_s emb[story[b,m0+r,s]][e] * pe[s,e]   (gather + VALU FMA)
//   ek[b,m0+r,h] = summed @ A_w^T                           (WMMA f16->f32)
// ---------------------------------------------------------------------------
__global__ __launch_bounds__(256)
void ek_kernel(const int* __restrict__ story, const float* __restrict__ emb,
               const float* __restrict__ A_w, const float* __restrict__ pe,
               float* __restrict__ ek)
{
    __shared__ _Float16 sum16[16][NEMBD];   // 16x128 f16 A tile
    __shared__ _Float16 aw16[48][NEMBD];    // A_w rows (HID padded to 48), f16

    const int t  = threadIdx.x;
    const int b  = blockIdx.y;
    const int m0 = blockIdx.x * 16;

    // stage A_w into LDS as f16 (rows >= HID zeroed so padded N-tile is clean)
    for (int i = t; i < 48 * NEMBD; i += 256) {
        int h = i >> 7, e = i & 127;
        aw16[h][e] = (h < HID) ? (_Float16)A_w[h * NEMBD + e] : (_Float16)0.0f;
    }

    // gather + position-weighted sum: 16 threads per memory row, 8 e's each
    {
        const int l = t & 15;          // e-lane within row
        const int r = t >> 4;          // memory row within tile
        float acc[8] = {0.f,0.f,0.f,0.f,0.f,0.f,0.f,0.f};
        const int* srow = story + ((size_t)b * MEMN + m0 + r) * SENTN;
        for (int s = 0; s < SENTN; ++s) {
            const int   idx  = srow[s];
            const float nz   = (idx != 0) ? 1.0f : 0.0f;   // PAD row forced to 0
            const float* erow = emb + (size_t)idx * NEMBD;
            const float* prow = pe + s * NEMBD;
            #pragma unroll
            for (int j = 0; j < 8; ++j) {
                int e = l + 16 * j;                        // coalesced across lanes
                acc[j] = fmaf(nz * erow[e], prow[e], acc[j]);
            }
        }
        #pragma unroll
        for (int j = 0; j < 8; ++j) sum16[r][l + 16 * j] = (_Float16)acc[j];
    }
    __syncthreads();

    // WMMA projection: D(16x16) += A(16x32) x B(32x16), K = 128 in 4 steps.
    // Waves 0..2 each own one 16-wide N-tile of HID (padded to 48).
    const int wave = t >> 5;
    const int lane = t & 31;
    if (wave < 3) {
        const int g  = lane >> 4;     // lane group (0/1)
        const int mn = lane & 15;     // M for A/D, N for B/D
        const int n0 = wave * 16;
        v8f c = {};
        #pragma unroll
        for (int kk = 0; kk < NEMBD; kk += 32) {
            v16h a, bf;
            #pragma unroll
            for (int e = 0; e < 16; ++e) {
                // ISA 7.12.2 16-bit A 16x32 layout
                int ka = (e < 8) ? (g * 8 + e) : (16 + g * 8 + (e - 8));
                a[e]  = sum16[mn][kk + ka];
                // B 32x16 layout: same K order as A (lanes 0-15 K=0..7,16..23; lanes 16-31 K=8..15,24..31)
                bf[e] = aw16[n0 + mn][kk + ka];     // B[k][n] = A_w[n][k]
            }
            c = __builtin_amdgcn_wmma_f32_16x16x32_f16(
                    false, a, false, bf, (short)0, c, false, false);
        }
        #pragma unroll
        for (int r = 0; r < 8; ++r) {                       // D: M = r + 8g, N = mn
            int n = n0 + mn;
            if (n < HID)
                ek[((size_t)b * MEMN + m0 + r + 8 * g) * HID + n] = c[r];
        }
    }
}

// ---------------------------------------------------------------------------
// Kernel 2: question encoding  eq[b,h] = (sum_s emb[q[b,s]]*pe[s]) @ A_w^T
// ---------------------------------------------------------------------------
__global__ __launch_bounds__(128)
void eq_kernel(const int* __restrict__ question, const float* __restrict__ emb,
               const float* __restrict__ A_w, const float* __restrict__ pe,
               float* __restrict__ eq)
{
    __shared__ float sq[NEMBD];
    const int b = blockIdx.x, e = threadIdx.x;
    float a = 0.f;
    for (int s = 0; s < SENTN; ++s) {
        int   idx = question[b * SENTN + s];
        float nz  = (idx != 0) ? 1.f : 0.f;
        a = fmaf(nz * emb[(size_t)idx * NEMBD + e], pe[s * NEMBD + e], a);
    }
    sq[e] = a;
    __syncthreads();
    if (e < HID) {
        float d = 0.f;
        for (int k = 0; k < NEMBD; ++k) d = fmaf(sq[k], A_w[e * NEMBD + k], d);
        eq[b * HID + e] = d;
    }
}

// ---------------------------------------------------------------------------
// Kernel 3: attention hops + readout + log_softmax, one block per batch item
// ---------------------------------------------------------------------------
__global__ __launch_bounds__(256)
void attn_kernel(const int* __restrict__ story, const float* __restrict__ ek,
                 const float* __restrict__ eq, const float* __restrict__ Rs,
                 const float* __restrict__ Wd, const float* __restrict__ bd,
                 float* __restrict__ out)
{
    __shared__ float q[HID], o[HID], tq[HID];
    __shared__ float sc[MEMN];
    __shared__ float red[256];
    __shared__ float lg[NANS];
    __shared__ float stats[2];

    const int b = blockIdx.x, t = threadIdx.x;
    const float* ekb = ek + (size_t)b * MEMN * HID;

    if (t < HID) q[t] = eq[b * HID + t];
    __syncthreads();

    // sequence: o=attend(q); q=(q+o)R_i^T  for i=0..2 (final attend is dead)
    for (int i = 0; i < HOPSN; ++i) {
        // scores + mask
        for (int m = t; m < MEMN; m += 256) {
            float d = 0.f;
            const float* er = ekb + (size_t)m * HID;
            for (int h = 0; h < HID; ++h) d = fmaf(q[h], er[h], d);
            if (story[((size_t)b * MEMN + m) * SENTN] == 0) d = NEGV;
            sc[m] = d;
        }
        __syncthreads();
        // softmax: max
        float lmax = -3.0e38f;
        for (int m = t; m < MEMN; m += 256) lmax = fmaxf(lmax, sc[m]);
        red[t] = lmax; __syncthreads();
        for (int s2 = 128; s2 > 0; s2 >>= 1) {
            if (t < s2) red[t] = fmaxf(red[t], red[t + s2]);
            __syncthreads();
        }
        const float mx = red[0]; __syncthreads();
        // softmax: exp + sum
        float lsum = 0.f;
        for (int m = t; m < MEMN; m += 256) {
            float v = __expf(sc[m] - mx);
            sc[m] = v;
            lsum += v;
        }
        red[t] = lsum; __syncthreads();
        for (int s2 = 128; s2 > 0; s2 >>= 1) {
            if (t < s2) red[t] += red[t + s2];
            __syncthreads();
        }
        const float inv = 1.f / red[0];
        __syncthreads();
        // o = attn @ ev ; tq = q + o
        if (t < HID) {
            float a = 0.f;
            for (int m = 0; m < MEMN; ++m)
                a = fmaf(sc[m], ekb[(size_t)m * HID + t], a);
            o[t]  = a * inv;
            tq[t] = q[t] + o[t];
        }
        __syncthreads();
        // q = tq @ Rs[i]^T
        if (t < HID) {
            const float* R = Rs + ((size_t)i * HID + t) * HID;
            float a = 0.f;
            for (int j = 0; j < HID; ++j) a = fmaf(tq[j], R[j], a);
            q[t] = a;
        }
        __syncthreads();
    }

    // logits + log_softmax
    if (t < NANS) {
        const float* W = Wd + t * HID;
        float a = bd[t];
        for (int h = 0; h < HID; ++h) a = fmaf(q[h], W[h], a);
        lg[t] = a;
    }
    __syncthreads();
    if (t == 0) {
        float mx = -3.0e38f;
        for (int a2 = 0; a2 < NANS; ++a2) mx = fmaxf(mx, lg[a2]);
        float s = 0.f;
        for (int a2 = 0; a2 < NANS; ++a2) s += __expf(lg[a2] - mx);
        stats[0] = mx;
        stats[1] = __logf(s);
    }
    __syncthreads();
    if (t < NANS) out[b * NANS + t] = lg[t] - stats[0] - stats[1];
}

// ---------------------------------------------------------------------------
extern "C" void kernel_launch(void* const* d_in, const int* in_sizes, int n_in,
                              void* d_out, int out_size, void* d_ws, size_t ws_size,
                              hipStream_t stream)
{
    (void)in_sizes; (void)n_in; (void)out_size; (void)ws_size;
    const int*   story    = (const int*)  d_in[0];
    const int*   question = (const int*)  d_in[1];
    // d_in[2] all_answers: dead in reference
    const float* emb      = (const float*)d_in[3];
    const float* A_w      = (const float*)d_in[4];
    // d_in[5] B_w: dead in reference
    const float* Rs       = (const float*)d_in[6];
    const float* Wd       = (const float*)d_in[7];
    const float* bd       = (const float*)d_in[8];
    const float* pe       = (const float*)d_in[9];
    float* out = (float*)d_out;

    float* ek = (float*)d_ws;                          // 64*1024*40 f32 = 10.5 MB
    float* eq = ek + (size_t)BATCH * MEMN * HID;       // 64*40 f32

    dim3 gA(MEMN / 16, BATCH);                         // 4096 blocks -> all WGPs
    ek_kernel<<<gA, 256, 0, stream>>>(story, emb, A_w, pe, ek);
    eq_kernel<<<BATCH, 128, 0, stream>>>(question, emb, A_w, pe, eq);
    attn_kernel<<<BATCH, 256, 0, stream>>>(story, ek, eq, Rs, Wd, bd, out);
}
